// VectorQuantizer_6511170421494
// MI455X (gfx1250) — compile-verified, matrix operand order checked by reading, not run on hardware
//
#include <hip/hip_runtime.h>

// ---------------------------------------------------------------------------
// VQ-VAE vector quantizer for MI455X (gfx1250, wave32, WMMA + async-to-LDS).
//
// S = codebook_tile(16x32 f16) x X^T_tile(32x16 f16), f32 accumulate, via
// v_wmma_f32_16x16x32_f16.  inputs[b] is (D=256, HW=1024) row-major, i.e.
// already X^T, so the B operand needs no transpose.
// argmin over codes uses score = ||c||^2 - 2*(c.x)  (x^2 is a per-pixel
// constant, irrelevant for argmin, re-added for the MSE losses).
//
// Codebook tiles are staged with GLOBAL_LOAD_ASYNC_TO_LDS_B128 into a
// double-buffered LDS tile: one barrier per tile, next tile's fetch fully
// overlapped with the current tile's 8 WMMAs.
// ---------------------------------------------------------------------------

typedef __attribute__((ext_vector_type(16))) _Float16 v16h;
typedef __attribute__((ext_vector_type(8)))  _Float16 v8h;
typedef __attribute__((ext_vector_type(8)))  float    v8f;

#define KCODES 1024
#define DDIM   256
#define HW     1024
#define BATCH  32
#define ELEMS  (BATCH * DDIM * HW)       // 8388608
#define OUT_LOSS ELEMS                   // q_loss, e_loss, vq_loss
#define OUT_IDX  (ELEMS + 3)             // 32768 indices (stored as float)
#define NTILES (KCODES / 16)             // 64 code tiles
#define APITCH (DDIM + 8)                // padded LDS row pitch (halfs)

// ---------------------------------------------------------------------------
// Kernel 0: codebook f32 -> f16 (row-major K x D) + per-code squared norms.
// ---------------------------------------------------------------------------
__global__ void vq_prep(const float* __restrict__ cb,
                        _Float16* __restrict__ cb16,
                        float* __restrict__ c2) {
  int k = blockIdx.x * blockDim.x + threadIdx.x;
  if (k >= KCODES) return;
  const float* row = cb + (size_t)k * DDIM;
  _Float16* row16 = cb16 + (size_t)k * DDIM;
  float s = 0.f;
#pragma unroll 8
  for (int j = 0; j < DDIM; ++j) {
    float v = row[j];
    s += v * v;
    row16[j] = (_Float16)v;
  }
  c2[k] = s;
}

// Async direct-to-LDS 16B copy (ASYNCcnt-tracked; no VGPR staging).
__device__ __forceinline__ void async_b128_to_lds(unsigned lds_off,
                                                  const void* gaddr) {
  asm volatile("global_load_async_to_lds_b128 %0, %1, off"
               :: "v"(lds_off), "v"(gaddr)
               : "memory");
}

__device__ __forceinline__ void wait_asynccnt0() {
  asm volatile("s_wait_asynccnt 0" ::: "memory");
}

// ---------------------------------------------------------------------------
// Kernel 1: per-pixel argmin over 1024 codes + quantized writeback + partial
// squared-error sums.  Grid: 256 WGs x 256 threads (8 waves).  Each wave owns
// 16 pixels of one batch; each WG owns 128 pixels.
// ---------------------------------------------------------------------------
__launch_bounds__(256, 1)
__global__ void vq_argmin(const float* __restrict__ x,     // (B,256,32,32) f32
                          const float* __restrict__ cb,    // f32 codebook (gather)
                          const _Float16* __restrict__ cb16,
                          const float* __restrict__ c2,
                          float* __restrict__ out,
                          float* __restrict__ partials) {
  // Double-buffered padded LDS tile: 2 x 16 codes x 256 halfs (pitch 264).
  __shared__ _Float16 sA[2][16][APITCH];

  const int tid  = threadIdx.x;
  const int w    = tid >> 5;          // wave id 0..7
  const int lane = tid & 31;
  const int b    = blockIdx.x >> 3;   // batch
  const int hw0  = (blockIdx.x & 7) * 128 + w * 16;  // this wave's pixel base
  const int p    = lane & 15;         // pixel within wave tile / code row
  const int khl  = (lane < 16) ? 0 : 8;  // K-half split per ISA 16-bit layouts

  // Cooperative-staging coordinates (per thread: 32 bytes of the 8KB tile).
  const int srow = tid >> 4;          // 0..15
  const int scol = (tid & 15) * 16;   // halfs

  const float* xb = x + (size_t)b * DDIM * HW;

  // ---- Phase 1: load this wave's X^T fragments (16 px x 256 d) into VGPRs.
  // B-operand layout (ISA 7.12.2, 16-bit, mirrored from A): lane n<16 holds
  // column n with K = 32c + {0..7, 16..23}; lanes 16-31 hold K = {8..15,24..31}.
  v16h bfrag[8];
  float x2 = 0.f;
#pragma unroll
  for (int c = 0; c < 8; ++c) {
#pragma unroll
    for (int j = 0; j < 16; ++j) {
      const int d = 32 * c + khl + (j < 8 ? j : 8 + j);
      const float v = xb[(size_t)d * HW + hw0 + p];
      x2 += v * v;
      bfrag[c][j] = (_Float16)v;
    }
  }

  // ---- Prologue: async-stage tile 0 into buffer 0.
  {
    const _Float16* src = cb16 + (size_t)srow * DDIM + scol;
    const unsigned l0 = (unsigned)(size_t)&sA[0][srow][scol];
    async_b128_to_lds(l0, src);
    async_b128_to_lds(l0 + 16, src + 8);
  }
  wait_asynccnt0();
  __syncthreads();

  float minv = 3.0e38f;
  int   mini = 0;

  // ---- Phase 2: loop over 64 code tiles (double-buffered, 1 barrier/tile).
  for (int ct = 0; ct < NTILES; ++ct) {
    // Issue async stage of the next tile into the other buffer.
    if (ct + 1 < NTILES) {
      const _Float16* src =
          cb16 + (size_t)((ct + 1) * 16 + srow) * DDIM + scol;
      const unsigned l0 = (unsigned)(size_t)&sA[(ct + 1) & 1][srow][scol];
      async_b128_to_lds(l0, src);
      async_b128_to_lds(l0 + 16, src + 8);
    }
    if (ct + 2 < NTILES)
      __builtin_prefetch(cb16 + (size_t)(ct + 2) * 16 * DDIM + ((tid & 63) << 6), 0, 1);

    // 8 chained WMMAs accumulate c.x over D=256 from the current buffer.
    const _Float16 (*cur)[APITCH] = sA[ct & 1];
    v8f acc = {};
#pragma unroll
    for (int c = 0; c < 8; ++c) {
      // A-operand layout (ISA 7.12.2): lane m<16 holds row m, K = 32c+{0..7},
      // then K = 32c+{16..23}; lanes 16-31 hold K = {8..15} and {24..31}.
      const v8h lo = *reinterpret_cast<const v8h*>(&cur[p][32 * c + khl]);
      const v8h hi = *reinterpret_cast<const v8h*>(&cur[p][32 * c + khl + 16]);
      v16h a;
#pragma unroll
      for (int j = 0; j < 8; ++j) { a[j] = lo[j]; a[8 + j] = hi[j]; }
      acc = __builtin_amdgcn_wmma_f32_16x16x32_f16(
          /*neg_a=*/false, a, /*neg_b=*/false, bfrag[c],
          /*c_mod=*/(short)0, acc, /*reuse_a=*/false, /*reuse_b=*/false);
    }

    // C/D layout: VGPR r -> code M = r (lanes 0-15) or r+8 (lanes 16-31),
    // N = lane&15 = pixel.  score = ||c||^2 - 2*(c.x).
    const int mbase = ct * 16 + ((lane < 16) ? 0 : 8);
#pragma unroll
    for (int r = 0; r < 8; ++r) {
      const float s = __builtin_fmaf(-2.f, acc[r], c2[mbase + r]);
      if (s < minv) { minv = s; mini = mbase + r; }  // strict < keeps lowest idx
    }

    if (ct + 1 < NTILES) {
      wait_asynccnt0();   // next buffer fully landed in LDS
      __syncthreads();    // all waves done reading current buffer
    }
  }

  // ---- Phase 3: combine lane p (codes m%16<8) with lane p+16 (codes m%16>=8).
  {
    const float ov = __shfl_xor(minv, 16, 32);
    const int   oi = __shfl_xor(mini, 16, 32);
    if (ov < minv || (ov == minv && oi < mini)) { minv = ov; mini = oi; }
    x2 += __shfl_xor(x2, 16, 32);   // both halves of d covered across the pair
  }

  // ||x - c_min||^2 for this lane's pixel (f32 throughout).
  float perr = x2 + minv;

  // idx output (lanes 0-15 carry pixels 0-15 of this wave's strip).
  if (lane < 16)
    out[OUT_IDX + (size_t)b * HW + hw0 + p] = (float)mini;

  // ---- Phase 4: quantized writeback, q[b, :, pix] = codebook_f32[idx].
  // lane = p + 16*dhalf: 16 consecutive pixels per half -> coalesced HBM stores.
  {
    const int dhalf = lane >> 4;
    const float* crow = cb + (size_t)mini * DDIM;
    float* ob = out + (size_t)b * DDIM * HW + hw0 + p;
#pragma unroll 4
    for (int d0 = 0; d0 < DDIM; d0 += 2) {
      const int dd = d0 + dhalf;
      ob[(size_t)dd * HW] = crow[dd];
    }
  }

  // ---- Phase 5: deterministic loss partials (one slot per wave).
#pragma unroll
  for (int m = 1; m < 16; m <<= 1) perr += __shfl_xor(perr, m, 32);
  if (lane == 0) partials[blockIdx.x * 8 + w] = perr;
}

// ---------------------------------------------------------------------------
// Kernel 2: deterministic final reduction of 2048 partials -> 3 scalars.
// q_loss == e_loss == mse; vq_loss = mse * (1 + BETA), BETA = 0.25.
// ---------------------------------------------------------------------------
__global__ void vq_finalize(const float* __restrict__ partials,
                            float* __restrict__ out) {
  if (threadIdx.x != 0 || blockIdx.x != 0) return;
  float s = 0.f;
  for (int i = 0; i < 2048; ++i) s += partials[i];
  const float mse = s / (float)ELEMS;
  out[OUT_LOSS + 0] = mse;
  out[OUT_LOSS + 1] = mse;
  out[OUT_LOSS + 2] = mse * 1.25f;
}

// ---------------------------------------------------------------------------
extern "C" void kernel_launch(void* const* d_in, const int* in_sizes, int n_in,
                              void* d_out, int out_size, void* d_ws, size_t ws_size,
                              hipStream_t stream) {
  const float* x  = (const float*)d_in[0];   // (32,256,32,32) f32
  const float* cb = (const float*)d_in[1];   // (1024,256) f32
  float* out = (float*)d_out;

  // Workspace layout: [0, 512KB) f16 codebook | [512KB, +4KB) c2 | [+8KB) partials
  _Float16* cb16   = (_Float16*)d_ws;
  float*    c2     = (float*)((char*)d_ws + 524288);
  float*    parts  = (float*)((char*)d_ws + 524288 + 4096);

  vq_prep<<<4, 256, 0, stream>>>(cb, cb16, c2);
  vq_argmin<<<256, 256, 0, stream>>>(x, cb, cb16, c2, out, parts);
  vq_finalize<<<1, 1, 0, stream>>>(parts, out);
}
